// KMeans_45827301048341
// MI455X (gfx1250) — matrix operand order repaired and checked by reading, not compiled, not run on hardware
//
#include <hip/hip_runtime.h>
#include <cstdint>

#define BB 4
#define NN 100000
#define DD 64
#define KK 256
#define KM_ITERS 5

typedef __attribute__((ext_vector_type(16))) __bf16 v16bf;
typedef __attribute__((ext_vector_type(8)))  float  v8f;

__device__ __forceinline__ unsigned short f32_to_bf16_bits(float f) {
    unsigned int u = __builtin_bit_cast(unsigned int, f);
    unsigned int r = u + 0x7FFFu + ((u >> 16) & 1u);   // round-to-nearest-even
    return (unsigned short)(r >> 16);
}
__device__ __forceinline__ float bf16_bits_to_f32(unsigned short s) {
    unsigned int u = ((unsigned int)s) << 16;
    return __builtin_bit_cast(float, u);
}
__device__ __forceinline__ __bf16 bits_to_bf16(unsigned short s) {
    return __builtin_bit_cast(__bf16, s);
}

// centroids[b,k,:] = points[b, idx[k], :]
__global__ void km_gather(const float* __restrict__ pts, const int* __restrict__ idx,
                          float* __restrict__ cen) {
    int t = blockIdx.x * blockDim.x + threadIdx.x;
    if (t >= BB * KK) return;
    int b = t / KK, k = t % KK;
    const float* p = pts + ((size_t)b * NN + (size_t)idx[k]) * DD;
    float* c = cen + (size_t)t * DD;
    #pragma unroll 4
    for (int d = 0; d < DD; ++d) c[d] = p[d];
}

// per (b,k): centroid f32 -> bf16 hi/lo split, c2 = ||c||^2, zero sums/counts
__global__ void km_prep(const float* __restrict__ cen,
                        unsigned short* __restrict__ chi, unsigned short* __restrict__ clo,
                        float* __restrict__ c2, float* __restrict__ sums,
                        float* __restrict__ cnts) {
    int t = blockIdx.x * blockDim.x + threadIdx.x;
    if (t >= BB * KK) return;
    const float* c = cen + (size_t)t * DD;
    unsigned short* h = chi + (size_t)t * DD;
    unsigned short* l = clo + (size_t)t * DD;
    float* s = sums + (size_t)t * DD;
    float acc = 0.0f;
    #pragma unroll 4
    for (int d = 0; d < DD; ++d) {
        float x = c[d];
        unsigned short hb = f32_to_bf16_bits(x);
        float hf = bf16_bits_to_f32(hb);
        unsigned short lb = f32_to_bf16_bits(x - hf);
        h[d] = hb; l[d] = lb;
        acc = fmaf(x, x, acc);
        s[d] = 0.0f;
    }
    c2[t] = acc;
    cnts[t] = 0.0f;
}

// Fused assignment (bf16x3 compensated WMMA GEMM + argmin) and LDS-staged segment-sum.
__global__ __launch_bounds__(256) void
km_assign(const float* __restrict__ pts,
          const unsigned short* __restrict__ chi, const unsigned short* __restrict__ clo,
          const float* __restrict__ c2,
          float* __restrict__ gsums, float* __restrict__ gcnts,
          float* __restrict__ out_assign, int do_accum) {
    __shared__ float s_sums[KK * DD];   // 64 KB (320 KB/WGP available on CDNA5)
    __shared__ float s_cnt[KK];
    __shared__ float s_c2[KK];

    const int b      = blockIdx.y;
    const int tid    = threadIdx.x;
    const int lane   = tid & 31;
    const int wave   = tid >> 5;
    const int half16 = lane >> 4;       // 0: lanes 0-15, 1: lanes 16-31
    const int l15    = lane & 15;

    s_c2[tid] = c2[b * KK + tid];       // blockDim.x == KK == 256
    if (do_accum) {
        #pragma unroll
        for (int i = tid; i < KK * DD; i += 256) s_sums[i] = 0.0f;
        s_cnt[tid] = 0.0f;
    }
    __syncthreads();

    const int nTiles        = NN / 16;                    // 6250 exact
    const int wavesPerBatch = gridDim.x * 8;
    const int waveGlobal    = blockIdx.x * 8 + wave;

    for (int t = waveGlobal; t < nTiles; t += wavesPerBatch) {   // wave-uniform
        const int base = t * 16;
        const float* prow = pts + ((size_t)b * NN + base + l15) * DD;

        // ---- Load A tile (16x64 f32) and split into bf16 hi/lo, contiguous k = 16*half16 + e per 32-k chunk ----
        v16bf ahi[2], alo[2];
        #pragma unroll
        for (int c = 0; c < 2; ++c) {
            const int d0 = c * 32 + half16 * 16;
            float4 q0 = *(const float4*)(prow + d0);
            float4 q1 = *(const float4*)(prow + d0 + 4);
            float4 q2 = *(const float4*)(prow + d0 + 8);
            float4 q3 = *(const float4*)(prow + d0 + 12);
            float x[16] = {q0.x,q0.y,q0.z,q0.w, q1.x,q1.y,q1.z,q1.w,
                           q2.x,q2.y,q2.z,q2.w, q3.x,q3.y,q3.z,q3.w};
            #pragma unroll
            for (int e = 0; e < 16; ++e) {
                unsigned short hb = f32_to_bf16_bits(x[e]);
                float hf = bf16_bits_to_f32(hb);
                unsigned short lb = f32_to_bf16_bits(x[e] - hf);
                ahi[c][e] = bits_to_bf16(hb);
                alo[c][e] = bits_to_bf16(lb);
            }
        }

        float bestv[8]; int besti[8];
        #pragma unroll
        for (int r = 0; r < 8; ++r) { bestv[r] = 3.0e38f; besti[r] = 0; }

        // ---- 16 column tiles of centroids; 6 WMMAs each ----
        // Keep this loop ROLLED: full unroll spills to scratch (>256 VGPRs).
        #pragma unroll 1
        for (int tc = 0; tc < 16; ++tc) {
            const int col = tc * 16 + l15;
            const unsigned short* ph = chi + ((size_t)b * KK + col) * DD;
            const unsigned short* pl = clo + ((size_t)b * KK + col) * DD;
            v8f acc = {0.f,0.f,0.f,0.f,0.f,0.f,0.f,0.f};
            #pragma unroll
            for (int c = 0; c < 2; ++c) {
                const int off = c * 32 + half16 * 16;       // same contiguous k order as A: K 0-15 / 16-31
                v16bf bh = *(const v16bf*)(ph + off);
                v16bf bl = *(const v16bf*)(pl + off);
                acc = __builtin_amdgcn_wmma_f32_16x16x32_bf16(false, alo[c], false, bh,
                                                              (short)0, acc, false, false);
                acc = __builtin_amdgcn_wmma_f32_16x16x32_bf16(false, ahi[c], false, bl,
                                                              (short)0, acc, false, false);
                acc = __builtin_amdgcn_wmma_f32_16x16x32_bf16(false, ahi[c], false, bh,
                                                              (short)0, acc, false, false);
            }
            const float c2v = s_c2[col];
            #pragma unroll
            for (int r = 0; r < 8; ++r) {
                float dist = fmaf(-2.0f, acc[r], c2v);      // p2 is row-constant: dropped
                bool take = dist < bestv[r];
                bestv[r] = take ? dist : bestv[r];
                besti[r] = take ? col  : besti[r];
            }
        }

        // ---- branchless argmin across the 16 lanes sharing each row ----
        #pragma unroll
        for (int r = 0; r < 8; ++r) {
            #pragma unroll
            for (int m = 1; m < 16; m <<= 1) {
                float ov = __shfl_xor(bestv[r], m, 32);
                int   oi = __shfl_xor(besti[r], m, 32);
                // bitwise combine: no short-circuit -> no EXEC-mask branches
                int take = (int)(ov < bestv[r]) | ((int)(ov == bestv[r]) & (int)(oi < besti[r]));
                bestv[r] = take ? ov : bestv[r];
                besti[r] = take ? oi : besti[r];
            }
        }
        // lanes 0..15 now hold rows 0-7; lanes 16..31 hold rows 8-15

        if (out_assign != nullptr && l15 == 0) {
            #pragma unroll
            for (int r = 0; r < 8; ++r)
                out_assign[(size_t)b * NN + base + half16 * 8 + r] = (float)besti[r];
        }

        if (do_accum) {
            #pragma unroll
            for (int rr = 0; rr < 16; ++rr) {
                const int k = __shfl(besti[rr & 7], (rr < 8) ? 0 : 16, 32);
                const float2 pv = *(const float2*)(pts + ((size_t)b * NN + base + rr) * DD
                                                   + lane * 2);
                atomicAdd(&s_sums[k * DD + lane * 2],     pv.x);
                atomicAdd(&s_sums[k * DD + lane * 2 + 1], pv.y);
                if (lane == 0) atomicAdd(&s_cnt[k], 1.0f);
            }
        }
    }

    if (do_accum) {
        __syncthreads();
        for (int i = tid; i < KK * DD; i += 256)
            atomicAdd(&gsums[(size_t)b * KK * DD + i], s_sums[i]);
        for (int i = tid; i < KK; i += 256)
            atomicAdd(&gcnts[b * KK + i], s_cnt[i]);
    }
}

__global__ void km_update(float* __restrict__ cen, const float* __restrict__ sums,
                          const float* __restrict__ cnts) {
    int t = blockIdx.x * blockDim.x + threadIdx.x;
    if (t >= BB * KK) return;
    float inv = 1.0f / cnts[t];          // 0-count -> inf; 0*inf=NaN matches reference 0/0
    const float* s = sums + (size_t)t * DD;
    float* c = cen + (size_t)t * DD;
    #pragma unroll 4
    for (int d = 0; d < DD; ++d) c[d] = s[d] * inv;
}

__global__ void km_copyout(const float* __restrict__ cen, float* __restrict__ out) {
    int t = blockIdx.x * blockDim.x + threadIdx.x;
    if (t < BB * KK * DD) out[t] = cen[t];
}

extern "C" void kernel_launch(void* const* d_in, const int* in_sizes, int n_in,
                              void* d_out, int out_size, void* d_ws, size_t ws_size,
                              hipStream_t stream) {
    const float* pts = (const float*)d_in[0];
    const int*   idx = (const int*)d_in[1];     // JAX x64 off: init_idx lands as int32
    float* out = (float*)d_out;                 // [B*N assignments][B*K*D centroids]

    char* ws = (char*)d_ws;                     // ~780 KB total
    float*          cen  = (float*)(ws + 0);          // B*K*D f32   = 262144 B
    float*          sums = (float*)(ws + 262144);     // B*K*D f32   = 262144 B
    float*          cnts = (float*)(ws + 524288);     // B*K f32     =   4096 B
    float*          c2   = (float*)(ws + 528384);     // B*K f32     =   4096 B
    unsigned short* chi  = (unsigned short*)(ws + 532480);  // B*K*D bf16 = 131072 B
    unsigned short* clo  = (unsigned short*)(ws + 663552);  // B*K*D bf16 = 131072 B

    km_gather<<<dim3(4), dim3(256), 0, stream>>>(pts, idx, cen);

    for (int it = 0; it < KM_ITERS; ++it) {
        km_prep<<<dim3(4), dim3(256), 0, stream>>>(cen, chi, clo, c2, sums, cnts);
        km_assign<<<dim3(128, BB), dim3(256), 0, stream>>>(pts, chi, clo, c2,
                                                           sums, cnts, nullptr, 1);
        km_update<<<dim3(4), dim3(256), 0, stream>>>(cen, sums, cnts);
    }

    // final assignment with converged centroids
    km_prep<<<dim3(4), dim3(256), 0, stream>>>(cen, chi, clo, c2, sums, cnts);
    km_assign<<<dim3(128, BB), dim3(256), 0, stream>>>(pts, chi, clo, c2,
                                                       sums, cnts, out, 0);
    km_copyout<<<dim3(256), dim3(256), 0, stream>>>(cen, out + (size_t)BB * NN);
}
